// MultiHeadAttentionLayer_3040836845942
// MI455X (gfx1250) — matrix operand order repaired and checked by reading, not compiled, not run on hardware
//
#include <hip/hip_runtime.h>

// ---------------- problem constants ----------------
#define B_  2
#define T_  2048
#define D_  1024
#define H_  16
#define DH_ 64
#define BT_ (B_*T_)
#define EPS_ 1e-3f

// ---------------- vector types ----------------
typedef __attribute__((ext_vector_type(16))) _Float16 v16h;
typedef __attribute__((ext_vector_type(8)))  _Float16 h8;
typedef __attribute__((ext_vector_type(4)))  _Float16 h4;
typedef __attribute__((ext_vector_type(8)))  float    v8f;
typedef __attribute__((ext_vector_type(4)))  int      v4i;

// ---------------- async global->LDS copy (gfx1250 ASYNCcnt path) ----------------
#if defined(__AMDGCN__) && __has_builtin(__builtin_amdgcn_global_load_async_to_lds_b128) && \
    __has_builtin(__builtin_amdgcn_s_wait_asynccnt)
#define USE_ASYNC 1
#else
#define USE_ASYNC 0
#endif

#if USE_ASYNC
typedef __attribute__((address_space(1))) v4i gv4i;   // global int4
typedef __attribute__((address_space(3))) v4i lv4i;   // LDS int4
__device__ __forceinline__ void async_cp16(const _Float16* g, _Float16* l) {
  __builtin_amdgcn_global_load_async_to_lds_b128((gv4i*)g, (lv4i*)l, 0, 0);
}
__device__ __forceinline__ void async_wait0() {
  __builtin_amdgcn_s_wait_asynccnt(0);
}
#endif

__device__ __forceinline__ v16h make_v16(h8 lo, h8 hi) {
  return __builtin_shufflevector(lo, hi, 0,1,2,3,4,5,6,7,8,9,10,11,12,13,14,15);
}

// A-matrix fragment (16x32 f16, M = lane%16).
// lanes 0-15 hold K {0..7,16..23}, lanes 16-31 hold K {8..15,24..31}.
__device__ __forceinline__ v16h load_afrag(const _Float16* rowbase, int lane) {
  const int koff = (lane & 16) ? 8 : 0;
  h8 lo = *(const h8*)(rowbase + koff);
  h8 hi = *(const h8*)(rowbase + 16 + koff);
  return make_v16(lo, hi);
}

// B-matrix fragment (32x16 f16): N = lane%16; lanes 0-15 hold K {0..7,16..23}, lanes 16-31 hold K {8..15,24..31}.
__device__ __forceinline__ v16h load_bfrag(const _Float16* p, int ld, int lane) {
  const int koff = (lane & 16) ? 8 : 0;
  v16h r;
#pragma unroll
  for (int s = 0; s < 8; ++s) {
    r[s]     = p[(koff + s) * ld];
    r[8 + s] = p[(16 + koff + s) * ld];
  }
  return r;
}

// ---------------- fp32 -> f16 conversion ----------------
__global__ void cvt_f32_f16(const float4* __restrict__ s, _Float16* __restrict__ d, int n4) {
  int i = blockIdx.x * blockDim.x + threadIdx.x;
  int stride = gridDim.x * blockDim.x;
  for (; i < n4; i += stride) {
    float4 v = s[i];
    h4 o;
    o.x = (_Float16)v.x; o.y = (_Float16)v.y;
    o.z = (_Float16)v.z; o.w = (_Float16)v.w;
    *(h4*)(d + (size_t)i * 4) = o;
  }
}

// ---------------- tiled WMMA GEMM: C = A(f16)*B(f16) + bias (+resid) ----------------
// 128x128x32 tiles, 256 threads = 8 waves; LDS double-buffered, async DMA staging.
#define BM 128
#define BN 128
#define BK 32
#define AS_STRIDE 40   // 32 + 8 pad (80B pitch, 16B aligned)
#define BS_STRIDE 136  // 128 + 8 pad (272B pitch, 16B aligned)

__global__ __launch_bounds__(256)
void gemm_bias(const _Float16* __restrict__ A, const _Float16* __restrict__ Bw,
               const float* __restrict__ bias, const float* __restrict__ resid,
               _Float16* __restrict__ outH, float* __restrict__ outF,
               int M, int N, int K)
{
  __shared__ __align__(16) _Float16 As[2][BM * AS_STRIDE];
  __shared__ __align__(16) _Float16 Bs[2][BK * BS_STRIDE];

  const int tid  = threadIdx.x;
  const int lane = tid & 31;
  const int wave = tid >> 5;
  const int m0 = blockIdx.y * BM;
  const int n0 = blockIdx.x * BN;

  // per-thread staging slots (two 8-half chunks per tile for A and for B)
  const int ar = tid >> 2, ac = (tid & 3) * 8;    // A: rows ar, ar+64
  const int br = tid >> 4, bc = (tid & 15) * 8;   // B: rows br, br+16
  const _Float16* gA = A + (size_t)(m0 + ar) * K + ac;
  const _Float16* gB = Bw + (size_t)br * N + n0 + bc;
  const size_t gAs = (size_t)64 * K;              // A chunk-1 offset
  const size_t gBs = (size_t)16 * N;              // B chunk-1 offset
  const int aoff = ar * AS_STRIDE + ac;
  const int boff = br * BS_STRIDE + bc;

  v8f acc[8];
#pragma unroll
  for (int n = 0; n < 8; ++n)
#pragma unroll
    for (int j = 0; j < 8; ++j) acc[n][j] = 0.0f;

  const int nT = K / BK;

#if USE_ASYNC
  // prologue: DMA tile 0 into buffer 0
  async_cp16(gA,       &As[0][aoff]);
  async_cp16(gA + gAs, &As[0][aoff + 64 * AS_STRIDE]);
  async_cp16(gB,       &Bs[0][boff]);
  async_cp16(gB + gBs, &Bs[0][boff + 16 * BS_STRIDE]);
  gA += BK; gB += (size_t)BK * N;
#else
  h8 ra0 = *(const h8*)gA, ra1 = *(const h8*)(gA + gAs);
  h8 rb0 = *(const h8*)gB, rb1 = *(const h8*)(gB + gBs);
  gA += BK; gB += (size_t)BK * N;
#endif

  for (int t = 0; t < nT; ++t) {
    const int cur = t & 1, nxt = cur ^ 1;
#if USE_ASYNC
    async_wait0();            // our DMA chunks for tile t have landed
    __syncthreads();          // everyone's chunks have landed
    if (t + 1 < nT) {         // overlap tile t+1 DMA with tile t math
      async_cp16(gA,       &As[nxt][aoff]);
      async_cp16(gA + gAs, &As[nxt][aoff + 64 * AS_STRIDE]);
      async_cp16(gB,       &Bs[nxt][boff]);
      async_cp16(gB + gBs, &Bs[nxt][boff + 16 * BS_STRIDE]);
      gA += BK; gB += (size_t)BK * N;
    }
#else
    *(h8*)(&As[cur][aoff]) = ra0;
    *(h8*)(&As[cur][aoff + 64 * AS_STRIDE]) = ra1;
    *(h8*)(&Bs[cur][boff]) = rb0;
    *(h8*)(&Bs[cur][boff + 16 * BS_STRIDE]) = rb1;
    if (t + 1 < nT) {         // issue tile t+1 loads; consumed next iteration
      ra0 = *(const h8*)gA; ra1 = *(const h8*)(gA + gAs);
      rb0 = *(const h8*)gB; rb1 = *(const h8*)(gB + gBs);
      gA += BK; gB += (size_t)BK * N;
    }
    __syncthreads();
#endif

    v16h af = load_afrag(&As[cur][(wave * 16 + (lane & 15)) * AS_STRIDE], lane);
#pragma unroll
    for (int ns = 0; ns < 8; ++ns) {
      v16h bf = load_bfrag(&Bs[cur][ns * 16 + (lane & 15)], BS_STRIDE, lane);
      acc[ns] = __builtin_amdgcn_wmma_f32_16x16x32_f16(
          false, af, false, bf, (short)0, acc[ns], false, false);
    }
    __syncthreads();          // all reads of buffer `cur` done before it is refilled
  }

  // epilogue: row = m0 + wave*16 + j + 8*(lane>=16), col = n0 + ns*16 + lane%16
  const int rowoff = (lane >> 4) << 3;
#pragma unroll
  for (int ns = 0; ns < 8; ++ns) {
    int col = n0 + ns * 16 + (lane & 15);
    float bv = bias[col];
#pragma unroll
    for (int j = 0; j < 8; ++j) {
      int row = m0 + wave * 16 + rowoff + j;
      size_t idx = (size_t)row * N + col;
      float v = acc[ns][j] + bv;
      if (resid) v += resid[idx];
      if (outH) outH[idx] = (_Float16)v;
      else      outF[idx] = v;
    }
  }
}

// ---------------- flash-style attention ----------------
// Grid: (T/64, H, B). 128 threads = 4 waves; each wave owns 16 query rows.
#define KT  32
#define VSS 72   // V tile row pitch (64+8)
#define KTT 40   // transposed-K tile row pitch (32+8)
#define PSS 40   // P scratch row pitch

__global__ __launch_bounds__(128)
void attention(const _Float16* __restrict__ Qh, const _Float16* __restrict__ Kh,
               const _Float16* __restrict__ Vh, const float* __restrict__ scale_p,
               _Float16* __restrict__ Ctx)
{
  __shared__ __align__(16) _Float16 Kts[64 * KTT];      // [kd][key]  (K transposed)
  __shared__ __align__(16) _Float16 Vs [KT * VSS];      // [key][dh]
  __shared__ __align__(16) _Float16 Ps [4 * 16 * PSS];  // per-wave P (16x32)

  const int tid  = threadIdx.x;
  const int lane = tid & 31;
  const int wave = tid >> 5;
  const int h  = blockIdx.y;
  const int b  = blockIdx.z;
  const int q0 = blockIdx.x * 64 + wave * 16;
  const int hc = h * DH_;
  const float scl = scale_p[0];

  // persistent Q fragments (K = 0..31 and 32..63 of DH)
  const _Float16* qrow = Qh + (size_t)(b * T_ + q0 + (lane & 15)) * D_ + hc;
  const int koff = (lane & 16) ? 8 : 0;
  v16h qa0 = make_v16(*(const h8*)(qrow + koff),      *(const h8*)(qrow + 16 + koff));
  v16h qa1 = make_v16(*(const h8*)(qrow + 32 + koff), *(const h8*)(qrow + 48 + koff));

  // per-thread K/V staging pointers (chunk = 8 halves; chunk1 = +16 rows)
  const int vr = tid >> 3, vc = (tid & 7) * 8;
  const _Float16* gV = Vh + (size_t)(b * T_ + vr) * D_ + hc + vc;
  const _Float16* gK = Kh + (size_t)(b * T_ + vr) * D_ + hc + vc;
  const size_t cstep = (size_t)16 * D_;   // chunk-1 offset
  const size_t tstep = (size_t)KT * D_;   // per key tile
  _Float16* sV = &Vs[vr * VSS + vc];

  v8f o[4];
  float mrow[8], lrow[8];
#pragma unroll
  for (int n = 0; n < 4; ++n)
#pragma unroll
    for (int j = 0; j < 8; ++j) o[n][j] = 0.0f;
#pragma unroll
  for (int j = 0; j < 8; ++j) { mrow[j] = -3.0e38f; lrow[j] = 0.0f; }

  for (int kt = 0; kt < T_ / KT; ++kt) {
    __syncthreads();
#if USE_ASYNC
    async_cp16(gV, sV);
    async_cp16(gV + cstep, sV + 16 * VSS);
    h8 k0 = *(const h8*)gK;
    h8 k1 = *(const h8*)(gK + cstep);
#pragma unroll
    for (int e = 0; e < 8; ++e) {
      Kts[(vc + e) * KTT + vr]      = k0[e];
      Kts[(vc + e) * KTT + vr + 16] = k1[e];
    }
    async_wait0();
#else
    h8 vv0 = *(const h8*)gV;
    h8 vv1 = *(const h8*)(gV + cstep);
    h8 k0  = *(const h8*)gK;
    h8 k1  = *(const h8*)(gK + cstep);
    *(h8*)(sV) = vv0;
    *(h8*)(sV + 16 * VSS) = vv1;
#pragma unroll
    for (int e = 0; e < 8; ++e) {
      Kts[(vc + e) * KTT + vr]      = k0[e];
      Kts[(vc + e) * KTT + vr + 16] = k1[e];
    }
#endif
    if (kt + 1 < T_ / KT) {
      __builtin_prefetch(gV + tstep, 0, 0);
      __builtin_prefetch(gK + tstep, 0, 0);
    }
    gV += tstep; gK += tstep;
    __syncthreads();

    // S = Q * K^T  (two 16-key subtiles, K-dim 64 = 2 chained WMMA each)
    v8f s[2];
#pragma unroll
    for (int ns = 0; ns < 2; ++ns) {
      v8f z;
#pragma unroll
      for (int j = 0; j < 8; ++j) z[j] = 0.0f;
      v16h b0 = load_bfrag(&Kts[ns * 16 + (lane & 15)], KTT, lane);          // kd 0..31
      v16h b1 = load_bfrag(&Kts[32 * KTT + ns * 16 + (lane & 15)], KTT, lane);   // kd 32..63
      z = __builtin_amdgcn_wmma_f32_16x16x32_f16(false, qa0, false, b0, (short)0, z, false, false);
      z = __builtin_amdgcn_wmma_f32_16x16x32_f16(false, qa1, false, b1, (short)0, z, false, false);
      s[ns] = z;
    }

    // online softmax update (rows owned: j + 8*(lane/16); cols = lane%16 per subtile)
    float alpha[8];
#pragma unroll
    for (int j = 0; j < 8; ++j) {
      float v0 = s[0][j] * scl, v1 = s[1][j] * scl;
      float mx = fmaxf(v0, v1);
#pragma unroll
      for (int off = 1; off <= 8; off <<= 1) mx = fmaxf(mx, __shfl_xor(mx, off));
      float mn = fmaxf(mrow[j], mx);
      float a  = __expf(mrow[j] - mn);
      float p0 = __expf(v0 - mn), p1 = __expf(v1 - mn);
      float rs = p0 + p1;
#pragma unroll
      for (int off = 1; off <= 8; off <<= 1) rs += __shfl_xor(rs, off);
      lrow[j] = lrow[j] * a + rs;
      mrow[j] = mn;
      alpha[j] = a;
      int pidx = (wave * 16 + ((lane >> 4) << 3) + j) * PSS + (lane & 15);
      Ps[pidx]      = (_Float16)p0;
      Ps[pidx + 16] = (_Float16)p1;
    }
#pragma unroll
    for (int n = 0; n < 4; ++n)
#pragma unroll
      for (int j = 0; j < 8; ++j) o[n][j] *= alpha[j];

    // O += P * V   (P 16x32 as A-fragment; V 32x64 = 4 B-subtiles)
    v16h pa = load_afrag(&Ps[(wave * 16 + (lane & 15)) * PSS], lane);
#pragma unroll
    for (int n = 0; n < 4; ++n) {
      v16h bf = load_bfrag(&Vs[n * 16 + (lane & 15)], VSS, lane);
      o[n] = __builtin_amdgcn_wmma_f32_16x16x32_f16(
          false, pa, false, bf, (short)0, o[n], false, false);
    }
  }

  // write normalized context into merged [BT, D] layout (head = column block)
#pragma unroll
  for (int n = 0; n < 4; ++n)
#pragma unroll
    for (int j = 0; j < 8; ++j) {
      int row = b * T_ + q0 + ((lane >> 4) << 3) + j;
      int col = hc + n * 16 + (lane & 15);
      Ctx[(size_t)row * D_ + col] = (_Float16)(o[n][j] / lrow[j]);
    }
}

// ---------------- row LayerNorm ----------------
__global__ __launch_bounds__(256)
void layernorm_rows(const float* __restrict__ in, const float* __restrict__ gamma,
                    const float* __restrict__ beta, float* __restrict__ out)
{
  __shared__ float red[256];
  const int row = blockIdx.x;
  const int tid = threadIdx.x;
  const float* p = in + (size_t)row * D_;

  float v[4];
  float s = 0.0f;
#pragma unroll
  for (int i = 0; i < 4; ++i) { v[i] = p[tid + i * 256]; s += v[i]; }
  red[tid] = s; __syncthreads();
  for (int w = 128; w > 0; w >>= 1) {
    if (tid < w) red[tid] += red[tid + w];
    __syncthreads();
  }
  float mean = red[0] * (1.0f / D_);
  __syncthreads();

  float sq = 0.0f;
#pragma unroll
  for (int i = 0; i < 4; ++i) { float d = v[i] - mean; sq += d * d; }
  red[tid] = sq; __syncthreads();
  for (int w = 128; w > 0; w >>= 1) {
    if (tid < w) red[tid] += red[tid + w];
    __syncthreads();
  }
  float var = red[0] * (1.0f / D_);
  float rs = rsqrtf(var + EPS_);
#pragma unroll
  for (int i = 0; i < 4; ++i) {
    int c = tid + i * 256;
    out[(size_t)row * D_ + c] = (v[i] - mean) * rs * gamma[c] + beta[c];
  }
}

// ---------------- launcher ----------------
extern "C" void kernel_launch(void* const* d_in, const int* in_sizes, int n_in,
                              void* d_out, int out_size, void* d_ws, size_t ws_size,
                              hipStream_t stream) {
  const float* x     = (const float*)d_in[0];
  const float* wq    = (const float*)d_in[1];
  const float* bq    = (const float*)d_in[2];
  const float* wk    = (const float*)d_in[3];
  const float* bk    = (const float*)d_in[4];
  const float* wv    = (const float*)d_in[5];
  const float* bv    = (const float*)d_in[6];
  const float* wo    = (const float*)d_in[7];
  const float* bo    = (const float*)d_in[8];
  const float* scale = (const float*)d_in[9];
  const float* gamma = (const float*)d_in[10];
  const float* beta  = (const float*)d_in[11];
  float* out = (float*)d_out;

  // workspace layout (64 MB total)
  char* ws = (char*)d_ws;
  _Float16* xh  = (_Float16*)ws;                 ws += (size_t)BT_ * D_ * 2;  // 8 MB
  _Float16* wqh = (_Float16*)ws;                 ws += (size_t)D_ * D_ * 2;   // 2 MB
  _Float16* wkh = (_Float16*)ws;                 ws += (size_t)D_ * D_ * 2;
  _Float16* wvh = (_Float16*)ws;                 ws += (size_t)D_ * D_ * 2;
  _Float16* woh = (_Float16*)ws;                 ws += (size_t)D_ * D_ * 2;
  _Float16* Qh  = (_Float16*)ws;                 ws += (size_t)BT_ * D_ * 2;
  _Float16* Kh  = (_Float16*)ws;                 ws += (size_t)BT_ * D_ * 2;
  _Float16* Vh  = (_Float16*)ws;                 ws += (size_t)BT_ * D_ * 2;
  _Float16* Ctx = (_Float16*)ws;                 ws += (size_t)BT_ * D_ * 2;
  float*    tmp = (float*)ws;                                                // 16 MB

  // 1) convert inputs to f16
  cvt_f32_f16<<<2048, 256, 0, stream>>>((const float4*)x,  xh,  (BT_ * D_) / 4);
  cvt_f32_f16<<<1024, 256, 0, stream>>>((const float4*)wq, wqh, (D_ * D_) / 4);
  cvt_f32_f16<<<1024, 256, 0, stream>>>((const float4*)wk, wkh, (D_ * D_) / 4);
  cvt_f32_f16<<<1024, 256, 0, stream>>>((const float4*)wv, wvh, (D_ * D_) / 4);
  cvt_f32_f16<<<1024, 256, 0, stream>>>((const float4*)wo, woh, (D_ * D_) / 4);

  // 2) QKV projections
  dim3 gg(D_ / BN, BT_ / BM);
  gemm_bias<<<gg, 256, 0, stream>>>(xh, wqh, bq, nullptr, Qh, nullptr, BT_, D_, D_);
  gemm_bias<<<gg, 256, 0, stream>>>(xh, wkh, bk, nullptr, Kh, nullptr, BT_, D_, D_);
  gemm_bias<<<gg, 256, 0, stream>>>(xh, wvh, bv, nullptr, Vh, nullptr, BT_, D_, D_);

  // 3) attention (flash-style, per head/batch/64-row q tile)
  attention<<<dim3(T_ / 64, H_, B_), 128, 0, stream>>>(Qh, Kh, Vh, scale, Ctx);

  // 4) output projection + bias + residual -> f32 scratch
  gemm_bias<<<gg, 256, 0, stream>>>(Ctx, woh, bo, x, nullptr, tmp, BT_, D_, D_);

  // 5) LayerNorm -> d_out
  layernorm_rows<<<BT_, 256, 0, stream>>>(tmp, gamma, beta, out);
}
